// FermiNet_70841190580848
// MI455X (gfx1250) — matrix operand order repaired and checked by reading, not compiled, not run on hardware
//
#include <hip/hip_runtime.h>
#include <hip/hip_bf16.h>
#include <math.h>

#define NE   32
#define NU   16
#define NSV  256
#define NPV  32
#define HALFV 128
#define SIN  832     // 3*256 + 2*32
#define SIN0 64      // 56 padded to 64

typedef __attribute__((ext_vector_type(16))) _Float16 v16h;
typedef __attribute__((ext_vector_type(8)))  _Float16 v8h;
typedef __attribute__((ext_vector_type(8)))  float    v8f;

// ---- workspace layout (in halves), all weights stored transposed [N][Kpad] ----
#define OFF_WS0T 0u              // 256*64
#define OFF_WS1T 16384u          // 256*832
#define OFF_WS2T 229376u
#define OFF_WS3T 442368u
#define OFF_WUT  655360u         // 128*256
#define OFF_WDT  688128u
#define OFF_WWUT 720896u         // 16*128
#define OFF_WWDT 722944u

// fragment helpers -----------------------------------------------------------
__device__ __forceinline__ v16h frag_cat(const _Float16* p0, const _Float16* p1) {
  v8h lo = *(const v8h*)p0;
  v8h hi = *(const v8h*)p1;
  return __builtin_shufflevector(lo, hi, 0,1,2,3,4,5,6,7,8,9,10,11,12,13,14,15);
}
// A (16x32 f16, row-major [rows][ld]): lanes 0-15 row m hold K 0-7 & 16-23,
// lanes 16-31 hold K 8-15 & 24-31.
__device__ __forceinline__ v16h load_afrag(const _Float16* base, int ld, int row0, int kc, int lane) {
  int m  = row0 + (lane & 15);
  int kb = kc * 32 + ((lane & 16) ? 8 : 0);
  const _Float16* p = base + m * ld + kb;
  return frag_cat(p, p + 16);
}
// B (32x16 f16, weights transposed [N][ldK]): lanes 0-15 col n hold K 0-7 & 16-23,
// lanes 16-31 hold K 8-15 & 24-31.
__device__ __forceinline__ v16h load_bfrag(const _Float16* baseT, int ldK, int n0, int kc, int lane) {
  int n  = n0 + (lane & 15);
  int ks = kc * 32 + ((lane & 16) ? 8 : 0);
  const _Float16* p = baseT + n * ldK + ks;
  return frag_cat(p, p + 16);
}
__device__ __forceinline__ v8f wmma_f16(v16h a, v16h b, v8f c) {
  return __builtin_amdgcn_wmma_f32_16x16x32_f16(false, a, false, b, (short)0, c, false, false);
}

// prep: f32 [K][N] -> f16 transposed [N][Kpad] (zero padded) ------------------
__global__ void cvt_transpose(const float* __restrict__ src, _Float16* __restrict__ dst,
                              int K, int N, int Kpad) {
  int idx = blockIdx.x * blockDim.x + threadIdx.x;
  if (idx >= N * Kpad) return;
  int n = idx / Kpad, k = idx - n * Kpad;
  dst[idx] = (k < K) ? (_Float16)src[k * N + n] : (_Float16)0.0f;
}

// ---------------------------------------------------------------------------
struct __align__(16) Smem {
  alignas(16) _Float16 s_in[NE * SIN];      // 53248 B; reused as f16 copy of s_v later
  alignas(16) float sv[NE * NSV];           // 32768 B
  float su[NSV], sd[NSV];
  float pu0[NE * 4], pd0[NE * 4];
  float puL[3][NE * NPV], pdL[3][NE * NPV];
  float wp0[4 * NPV], wp1[NPV * NPV], wp2[NPV * NPV];
  float bp0v[NPV], bp1v[NPV], bp2v[NPV];
  float env[NE];
  float rl[NE * 3];
  float al[12];
  alignas(16) _Float16 suH[16 * HALFV];
  alignas(16) _Float16 sdH[16 * HALFV];
  float orbU[16 * 16], orbD[16 * 16];
  float facU[16], facD[16];
  int prU, prD;
};

__global__ __launch_bounds__(256) void ferminet_fwd(
    const float* __restrict__ r, const float* __restrict__ a,
    const float* __restrict__ bs0, const float* __restrict__ bs1,
    const float* __restrict__ bs2, const float* __restrict__ bs3,
    const float* __restrict__ Wp0, const float* __restrict__ bp0,
    const float* __restrict__ Wp1, const float* __restrict__ bp1,
    const float* __restrict__ Wp2, const float* __restrict__ bp2,
    const float* __restrict__ bu, const float* __restrict__ bdn,
    const float* __restrict__ bwu, const float* __restrict__ bwd,
    const _Float16* __restrict__ W, float* __restrict__ out) {
  __shared__ Smem s;
  const int t = threadIdx.x;
  const int lane = t & 31;
  const int wave = t >> 5;
  const int b = blockIdx.x;
  const float* rb = r + (size_t)b * NE * 3;

  // ---- Phase A: stage r, atoms, p-weights; zero mean accumulators ----
  if (t < NE * 3) s.rl[t] = rb[t];
  if (t < 12) s.al[t] = a[t];
  for (int i = t; i < NE * 4; i += 256) { s.pu0[i] = 0.f; s.pd0[i] = 0.f; }
  for (int i = t; i < 3 * NE * NPV; i += 256) {
    (&s.puL[0][0])[i] = 0.f; (&s.pdL[0][0])[i] = 0.f;
  }
  for (int i = t; i < 4 * NPV; i += 256) s.wp0[i] = Wp0[i];
  for (int i = t; i < NPV * NPV; i += 256) { s.wp1[i] = Wp1[i]; s.wp2[i] = Wp2[i]; }
  if (t < NPV) { s.bp0v[t] = bp0[t]; s.bp1v[t] = bp1[t]; s.bp2v[t] = bp2[t]; }
  __syncthreads();

  // one-electron features + envelopes
  if (t < NE) {
    float rx = s.rl[t * 3 + 0], ry = s.rl[t * 3 + 1], rz = s.rl[t * 3 + 2];
    float e = 0.f;
    #pragma unroll
    for (int aa = 0; aa < 4; ++aa) {
      float dx = rx - s.al[aa * 3 + 0];
      float dy = ry - s.al[aa * 3 + 1];
      float dz = rz - s.al[aa * 3 + 2];
      float len = sqrtf(dx * dx + dy * dy + dz * dz);
      s.sv[t * NSV + aa * 4 + 0] = dx;
      s.sv[t * NSV + aa * 4 + 1] = dy;
      s.sv[t * NSV + aa * 4 + 2] = dz;
      s.sv[t * NSV + aa * 4 + 3] = len;
      e += expf(-len);
    }
    s.env[t] = e;
  }
  __syncthreads();

  // ---- Phase B: roll all 1024 pair streams, accumulate pu/pd means ----
  for (int pair = t; pair < NE * NE; pair += 256) {
    int i = pair >> 5, j = pair & 31;
    float dx = s.rl[j * 3 + 0] - s.rl[i * 3 + 0];
    float dy = s.rl[j * 3 + 1] - s.rl[i * 3 + 1];
    float dz = s.rl[j * 3 + 2] - s.rl[i * 3 + 2];
    float len = (i == j) ? 0.f : sqrtf(dx * dx + dy * dy + dz * dz);
    float p0[4] = {dx, dy, dz, len};
    const bool up = (i < NU);
    float* A0 = up ? &s.pu0[j * 4] : &s.pd0[j * 4];
    #pragma unroll
    for (int d = 0; d < 4; ++d) atomicAdd(&A0[d], p0[d] * 0.0625f);
    float p[NPV], q[NPV];
    #pragma unroll
    for (int c = 0; c < NPV; ++c) {
      float acc = s.bp0v[c];
      #pragma unroll
      for (int d = 0; d < 4; ++d) acc += p0[d] * s.wp0[d * NPV + c];
      p[c] = tanhf(acc);
    }
    float* A1 = up ? &s.puL[0][j * NPV] : &s.pdL[0][j * NPV];
    #pragma unroll
    for (int c = 0; c < NPV; ++c) atomicAdd(&A1[c], p[c] * 0.0625f);
    #pragma unroll
    for (int c = 0; c < NPV; ++c) {
      float acc = s.bp1v[c];
      #pragma unroll
      for (int d = 0; d < NPV; ++d) acc += p[d] * s.wp1[d * NPV + c];
      q[c] = tanhf(acc) + p[c];
    }
    float* A2 = up ? &s.puL[1][j * NPV] : &s.pdL[1][j * NPV];
    #pragma unroll
    for (int c = 0; c < NPV; ++c) atomicAdd(&A2[c], q[c] * 0.0625f);
    #pragma unroll
    for (int c = 0; c < NPV; ++c) {
      float acc = s.bp2v[c];
      #pragma unroll
      for (int d = 0; d < NPV; ++d) acc += q[d] * s.wp2[d * NPV + c];
      p[c] = tanhf(acc) + q[c];
    }
    float* A3 = up ? &s.puL[2][j * NPV] : &s.pdL[2][j * NPV];
    #pragma unroll
    for (int c = 0; c < NPV; ++c) atomicAdd(&A3[c], p[c] * 0.0625f);
  }
  __syncthreads();

  // ---- Phase C: layer 0 (K padded 56->64) ----
  if (t < 16) {
    float acc = 0.f;
    for (int m = 0; m < 16; ++m) acc += s.sv[m * NSV + t];
    s.su[t] = acc * 0.0625f;
  } else if (t < 32) {
    int c = t - 16; float acc = 0.f;
    for (int m = 16; m < 32; ++m) acc += s.sv[m * NSV + c];
    s.sd[c] = acc * 0.0625f;
  }
  __syncthreads();
  for (int idx = t; idx < NE * SIN0; idx += 256) {
    int m = idx >> 6, c = idx & 63;
    float v;
    if (c < 16)       v = s.su[c];
    else if (c < 32)  v = s.sd[c - 16];
    else if (c < 36)  v = s.pu0[m * 4 + (c - 32)];
    else if (c < 40)  v = s.pd0[m * 4 + (c - 36)];
    else if (c < 56)  v = s.sv[m * NSV + (c - 40)];
    else              v = 0.f;
    s.s_in[idx] = (_Float16)v;
  }
  __syncthreads();
  {
    const _Float16* W0 = W + OFF_WS0T;
    #pragma unroll
    for (int ti = 0; ti < 4; ++ti) {
      int tile = wave + 8 * ti;
      int mt = tile & 1, nt = tile >> 1;
      v8f acc = {0.f, 0.f, 0.f, 0.f, 0.f, 0.f, 0.f, 0.f};
      #pragma unroll
      for (int kc = 0; kc < 2; ++kc)
        acc = wmma_f16(load_afrag(s.s_in, SIN0, mt * 16, kc, lane),
                       load_bfrag(W0, SIN0, nt * 16, kc, lane), acc);
      int n = nt * 16 + (lane & 15);
      int mb = mt * 16 + ((lane & 16) ? 8 : 0);
      float bias = bs0[n];
      #pragma unroll
      for (int r2 = 0; r2 < 8; ++r2)
        s.sv[(mb + r2) * NSV + n] = tanhf(acc[r2] + bias);
    }
  }
  __syncthreads();

  // ---- Phase D: layers 1..3 (832 -> 256 with residual) ----
  for (int l = 1; l <= 3; ++l) {
    const float* bsl = (l == 1) ? bs1 : (l == 2) ? bs2 : bs3;
    const _Float16* Wl = W + ((l == 1) ? OFF_WS1T : (l == 2) ? OFF_WS2T : OFF_WS3T);
    {
      float au = 0.f, ad = 0.f;
      for (int m = 0; m < 16; ++m) {
        au += s.sv[m * NSV + t];
        ad += s.sv[(m + 16) * NSV + t];
      }
      s.su[t] = au * 0.0625f;
      s.sd[t] = ad * 0.0625f;
    }
    __syncthreads();
    for (int idx = t; idx < NE * SIN; idx += 256) {
      int m = idx / SIN, c = idx - m * SIN;
      float v;
      if (c < 256)      v = s.su[c];
      else if (c < 512) v = s.sd[c - 256];
      else if (c < 544) v = s.puL[l - 1][m * NPV + (c - 512)];
      else if (c < 576) v = s.pdL[l - 1][m * NPV + (c - 544)];
      else              v = s.sv[m * NSV + (c - 576)];
      s.s_in[idx] = (_Float16)v;
    }
    __syncthreads();
    #pragma unroll
    for (int ti = 0; ti < 4; ++ti) {
      int tile = wave + 8 * ti;
      int mt = tile & 1, nt = tile >> 1;
      v8f acc = {0.f, 0.f, 0.f, 0.f, 0.f, 0.f, 0.f, 0.f};
      for (int kc = 0; kc < 26; ++kc)
        acc = wmma_f16(load_afrag(s.s_in, SIN, mt * 16, kc, lane),
                       load_bfrag(Wl, SIN, nt * 16, kc, lane), acc);
      int n = nt * 16 + (lane & 15);
      int mb = mt * 16 + ((lane & 16) ? 8 : 0);
      float bias = bsl[n];
      #pragma unroll
      for (int r2 = 0; r2 < 8; ++r2) {
        int m = mb + r2;
        float resid = (float)s.s_in[m * SIN + 576 + n];
        s.sv[m * NSV + n] = tanhf(acc[r2] + bias) + resid;
      }
    }
    __syncthreads();
  }

  // ---- Phase E: s_u/s_d (K=256 -> 128), then s_wu/s_wd (K=128 -> 16) ----
  _Float16* svH = s.s_in;  // reuse
  for (int i = t; i < NE * NSV; i += 256) svH[i] = (_Float16)s.sv[i];
  __syncthreads();
  #pragma unroll
  for (int ti = 0; ti < 2; ++ti) {
    int tile = wave * 2 + ti;           // 0..15
    int half = tile >> 3;               // 0 = up, 1 = down
    int nt = tile & 7;
    const _Float16* Wt = W + (half ? OFF_WDT : OFF_WUT);
    v8f acc = {0.f, 0.f, 0.f, 0.f, 0.f, 0.f, 0.f, 0.f};
    #pragma unroll
    for (int kc = 0; kc < 8; ++kc)
      acc = wmma_f16(load_afrag(svH, NSV, half * 16, kc, lane),
                     load_bfrag(Wt, NSV, nt * 16, kc, lane), acc);
    int n = nt * 16 + (lane & 15);
    int mb = (lane & 16) ? 8 : 0;
    const float* bv = half ? bdn : bu;
    _Float16* dst = half ? s.sdH : s.suH;
    float bias = bv[n];
    #pragma unroll
    for (int r2 = 0; r2 < 8; ++r2)
      dst[(mb + r2) * HALFV + n] = (_Float16)(acc[r2] + bias);
  }
  __syncthreads();
  if (wave < 2) {
    int half = wave;
    const _Float16* Wt = W + (half ? OFF_WWDT : OFF_WWUT);
    const _Float16* Ah = half ? s.sdH : s.suH;
    v8f acc = {0.f, 0.f, 0.f, 0.f, 0.f, 0.f, 0.f, 0.f};
    #pragma unroll
    for (int kc = 0; kc < 4; ++kc)
      acc = wmma_f16(load_afrag(Ah, HALFV, 0, kc, lane),
                     load_bfrag(Wt, HALFV, 0, kc, lane), acc);
    int n = lane & 15;
    int mb = (lane & 16) ? 8 : 0;
    const float* bw = half ? bwd : bwu;
    float* orb = half ? s.orbD : s.orbU;
    float bias = bw[n];
    #pragma unroll
    for (int r2 = 0; r2 < 8; ++r2) {
      int m = mb + r2;
      orb[m * 16 + n] = (acc[r2] + bias) * s.env[half * 16 + m];
    }
  }
  __syncthreads();

  // ---- Phase F: log|det| of two 16x16 matrices (partial-pivot LU) ----
  const int di = t >> 4, dj = t & 15;
  for (int k = 0; k < 16; ++k) {
    if (t == 0) {
      int pr = k; float best = fabsf(s.orbU[k * 16 + k]);
      for (int i2 = k + 1; i2 < 16; ++i2) {
        float v = fabsf(s.orbU[i2 * 16 + k]);
        if (v > best) { best = v; pr = i2; }
      }
      s.prU = pr;
    }
    if (t == 1) {
      int pr = k; float best = fabsf(s.orbD[k * 16 + k]);
      for (int i2 = k + 1; i2 < 16; ++i2) {
        float v = fabsf(s.orbD[i2 * 16 + k]);
        if (v > best) { best = v; pr = i2; }
      }
      s.prD = pr;
    }
    __syncthreads();
    if (di == k) {
      int pr = s.prU;
      if (pr != k) {
        float tmp = s.orbU[k * 16 + dj];
        s.orbU[k * 16 + dj] = s.orbU[pr * 16 + dj];
        s.orbU[pr * 16 + dj] = tmp;
      }
      pr = s.prD;
      if (pr != k) {
        float tmp = s.orbD[k * 16 + dj];
        s.orbD[k * 16 + dj] = s.orbD[pr * 16 + dj];
        s.orbD[pr * 16 + dj] = tmp;
      }
    }
    __syncthreads();
    if (dj == k && di > k) {
      s.facU[di] = s.orbU[di * 16 + k] / s.orbU[k * 16 + k];
      s.facD[di] = s.orbD[di * 16 + k] / s.orbD[k * 16 + k];
    }
    __syncthreads();
    if (di > k && dj > k) {
      s.orbU[di * 16 + dj] -= s.facU[di] * s.orbU[k * 16 + dj];
      s.orbD[di * 16 + dj] -= s.facD[di] * s.orbD[k * 16 + dj];
    }
    __syncthreads();
  }
  if (t == 0) {
    float ld = 0.f;
    for (int k = 0; k < 16; ++k)
      ld += logf(fabsf(s.orbU[k * 16 + k])) + logf(fabsf(s.orbD[k * 16 + k]));
    out[b] = ld;
  }
}

extern "C" void kernel_launch(void* const* d_in, const int* in_sizes, int n_in,
                              void* d_out, int out_size, void* d_ws, size_t ws_size,
                              hipStream_t stream) {
  const float* r   = (const float*)d_in[0];
  const float* a   = (const float*)d_in[1];
  const float* Ws0 = (const float*)d_in[2];
  const float* bs0 = (const float*)d_in[3];
  const float* Wp0 = (const float*)d_in[4];
  const float* bp0 = (const float*)d_in[5];
  const float* Ws1 = (const float*)d_in[6];
  const float* bs1 = (const float*)d_in[7];
  const float* Wp1 = (const float*)d_in[8];
  const float* bp1 = (const float*)d_in[9];
  const float* Ws2 = (const float*)d_in[10];
  const float* bs2 = (const float*)d_in[11];
  const float* Wp2 = (const float*)d_in[12];
  const float* bp2 = (const float*)d_in[13];
  const float* Ws3 = (const float*)d_in[14];
  const float* bs3 = (const float*)d_in[15];
  const float* Wu  = (const float*)d_in[16];
  const float* bu  = (const float*)d_in[17];
  const float* Wd  = (const float*)d_in[18];
  const float* bd  = (const float*)d_in[19];
  const float* Wwu = (const float*)d_in[20];
  const float* bwu = (const float*)d_in[21];
  const float* Wwd = (const float*)d_in[22];
  const float* bwd = (const float*)d_in[23];
  _Float16* W = (_Float16*)d_ws;

  auto T = [&](const float* src, unsigned off, int K, int N, int Kpad) {
    int total = N * Kpad;
    cvt_transpose<<<(total + 255) / 256, 256, 0, stream>>>(src, W + off, K, N, Kpad);
  };
  T(Ws0, OFF_WS0T, 56, 256, 64);
  T(Ws1, OFF_WS1T, 832, 256, 832);
  T(Ws2, OFF_WS2T, 832, 256, 832);
  T(Ws3, OFF_WS3T, 832, 256, 832);
  T(Wu,  OFF_WUT,  256, 128, 256);
  T(Wd,  OFF_WDT,  256, 128, 256);
  T(Wwu, OFF_WWUT, 128, 16, 128);
  T(Wwd, OFF_WWDT, 128, 16, 128);

  int nB = in_sizes[0] / (NE * 3);
  ferminet_fwd<<<nB, 256, 0, stream>>>(r, a, bs0, bs1, bs2, bs3,
                                       Wp0, bp0, Wp1, bp1, Wp2, bp2,
                                       bu, bd, bwu, bwd, W, (float*)d_out);
}
